// XDecoder_26723286515769
// MI455X (gfx1250) — compile-verified, matrix operand order checked by reading, not run on hardware
//
#include <hip/hip_runtime.h>
#include <math.h>

// ---------------- model constants ----------------
#define BDIM   8
#define NSEQ   256
#define ROWS   (BDIM * NSEQ)        // 2048
#define DIM    512
#define HEADS  8
#define DH     64
#define FFH    2048                 // FF_HID
#define FF2    (2 * FFH)            // 4096
#define EMBW   116
#define X0W    128                  // EMB + 12
#define NLAY   384

typedef __attribute__((ext_vector_type(16))) __bf16 v16bf;
typedef __attribute__((ext_vector_type(8)))  float  v8f;

// fp32 -> bf16, round-to-nearest-even (bit exact, no reliance on __bf16 arith)
static __device__ __forceinline__ __bf16 f2bf(float x) {
  unsigned u = __float_as_uint(x);
  unsigned r = (u + 0x7FFFu + ((u >> 16) & 1u)) >> 16;
  unsigned short s = (unsigned short)r;
  __bf16 b;
  __builtin_memcpy(&b, &s, 2);
  return b;
}

// ---------------- WMMA GEMM: C[M,N] = A[M,K] @ W[K,N] (+bias)(+residual) ----
// fp32 in memory, bf16 into the matrix pipe, fp32 accumulate.
// Block = 256 threads (8 wave32) -> 64x64 output tile.
//   wave w owns rows tm+16w..tm+16w+15 and all 64 columns (4 WMMA accumulators)
//   B panel (32k x 64n) is staged fp32->bf16 in LDS once per K-step and shared
//   by all 8 waves; A fragments come straight from global as b128 loads.
// Grids are exact so EXEC is all-ones at every v_wmma (ISA requirement).
#define BSTRIDE 34   // LDS row stride (bf16 elems) for B panel: even + non-pow2
__global__ void __launch_bounds__(256)
gemm_wmma_bf16(const float* __restrict__ A, const float* __restrict__ W,
               const float* __restrict__ bias, const float* __restrict__ res,
               float* __restrict__ C, int M, int N, int K) {
  __shared__ __bf16 Bs[64 * BSTRIDE];

  const int nb   = N >> 6;                 // 64-col blocks
  const int bm   = blockIdx.x / nb;
  const int bn   = blockIdx.x % nb;
  const int tnb  = bn << 6;                // first column of this block
  const int wave = threadIdx.x >> 5;
  const int lane = threadIdx.x & 31;
  const int g    = lane >> 4;              // lane group (0: lanes 0-15)
  const int l    = lane & 15;
  const int tm   = (bm << 6) + (wave << 4);// this wave's first row

  // staging coords: thread t loads 8 consecutive n for one k row
  const int skk = threadIdx.x >> 3;        // 0..31
  const int snn = (threadIdx.x & 7) << 3;  // 0,8,..,56

  v8f acc0 = {}, acc1 = {}, acc2 = {}, acc3 = {};

  for (int kb = 0; kb < K; kb += 32) {
    // ---- stage B panel (fp32 -> bf16) into LDS, [n][k] layout ----
    __syncthreads();
    {
      const float4* bp = reinterpret_cast<const float4*>(
          W + (size_t)(kb + skk) * N + tnb + snn);
      float4 b0 = bp[0];
      float4 b1 = bp[1];
      __bf16* d = &Bs[(size_t)snn * BSTRIDE + skk];
      d[0 * BSTRIDE] = f2bf(b0.x);
      d[1 * BSTRIDE] = f2bf(b0.y);
      d[2 * BSTRIDE] = f2bf(b0.z);
      d[3 * BSTRIDE] = f2bf(b0.w);
      d[4 * BSTRIDE] = f2bf(b1.x);
      d[5 * BSTRIDE] = f2bf(b1.y);
      d[6 * BSTRIDE] = f2bf(b1.z);
      d[7 * BSTRIDE] = f2bf(b1.w);
    }
    // prefetch next K panel into cache while we compute (global_prefetch_b8)
    if (kb + 32 < K) {
      __builtin_prefetch(W + (size_t)(kb + 32 + skk) * N + tnb + snn, 0, 1);
      __builtin_prefetch(A + (size_t)(tm + l) * K + kb + 32 + g * 8, 0, 1);
    }
    __syncthreads();

    // ---- A fragment: row tm+l; lane-group g holds K offs g*8..+7 (VGPR 0-3)
    //      and 16+g*8..+7 (VGPR 4-7), two bf16 per VGPR ----
    v16bf a;
    {
      const float4* ap = reinterpret_cast<const float4*>(
          A + (size_t)(tm + l) * K + kb + g * 8);
      float4 a0 = ap[0], a1 = ap[1];       // k rel 0..7
      float4 a2 = ap[4], a3 = ap[5];       // k rel 16..23
      a[0] = f2bf(a0.x); a[1] = f2bf(a0.y);
      a[2] = f2bf(a0.z); a[3] = f2bf(a0.w);
      a[4] = f2bf(a1.x); a[5] = f2bf(a1.y);
      a[6] = f2bf(a1.z); a[7] = f2bf(a1.w);
      a[8]  = f2bf(a2.x); a[9]  = f2bf(a2.y);
      a[10] = f2bf(a2.z); a[11] = f2bf(a2.w);
      a[12] = f2bf(a3.x); a[13] = f2bf(a3.y);
      a[14] = f2bf(a3.z); a[15] = f2bf(a3.w);
    }

    // ---- 4 WMMAs reuse the A fragment across the 64-wide strip ----
#define B_FRAG(j, bv)                                                   \
    {                                                                   \
      const __bf16* bnp = &Bs[(size_t)((j << 4) + l) * BSTRIDE + g * 8];\
      bv[0] = bnp[0];  bv[1] = bnp[1];  bv[2] = bnp[2];  bv[3] = bnp[3];\
      bv[4] = bnp[4];  bv[5] = bnp[5];  bv[6] = bnp[6];  bv[7] = bnp[7];\
      bv[8]  = bnp[16]; bv[9]  = bnp[17]; bv[10] = bnp[18];             \
      bv[11] = bnp[19]; bv[12] = bnp[20]; bv[13] = bnp[21];             \
      bv[14] = bnp[22]; bv[15] = bnp[23];                               \
    }
    v16bf b0, b1, b2, b3;
    B_FRAG(0, b0)
    B_FRAG(1, b1)
    B_FRAG(2, b2)
    B_FRAG(3, b3)
#undef B_FRAG
    acc0 = __builtin_amdgcn_wmma_f32_16x16x32_bf16(false, a, false, b0,
                                                   (short)0, acc0, false, false);
    acc1 = __builtin_amdgcn_wmma_f32_16x16x32_bf16(false, a, false, b1,
                                                   (short)0, acc1, false, false);
    acc2 = __builtin_amdgcn_wmma_f32_16x16x32_bf16(false, a, false, b2,
                                                   (short)0, acc2, false, false);
    acc3 = __builtin_amdgcn_wmma_f32_16x16x32_bf16(false, a, false, b3,
                                                   (short)0, acc3, false, false);
  }

  // ---- epilogue. C/D layout: lane group g, VGPR vv -> M = vv + 8g ; N = l.
  v8f* accs[4] = {&acc0, &acc1, &acc2, &acc3};
  if (res) {
#pragma unroll
    for (int j = 0; j < 4; ++j) {
      const int n = tnb + (j << 4) + l;
      const float bb = bias ? bias[n] : 0.0f;
#pragma unroll
      for (int vv = 0; vv < 8; ++vv) {
        const int m = tm + vv + g * 8;
        C[(size_t)m * N + n] = (*accs[j])[vv] + bb + res[(size_t)m * N + n];
      }
    }
  } else {
#pragma unroll
    for (int j = 0; j < 4; ++j) {
      const int n = tnb + (j << 4) + l;
      const float bb = bias ? bias[n] : 0.0f;
#pragma unroll
      for (int vv = 0; vv < 8; ++vv) {
        const int m = tm + vv + g * 8;
        C[(size_t)m * N + n] = (*accs[j])[vv] + bb;
      }
    }
  }
}

// ---------------- x0 = concat(emb[cls], feats) --------------------------
__global__ void build_x0(const float* __restrict__ saml,
                         const float* __restrict__ emb,
                         float* __restrict__ x0) {
  int idx = blockIdx.x * blockDim.x + threadIdx.x;   // ROWS*128
  int row = idx >> 7, c = idx & 127;
  const float* s = saml + (size_t)row * 13;
  float v;
  if (c < EMBW) {
    int cls = (int)s[0];
    v = emb[(size_t)cls * EMBW + c];
  } else {
    v = s[1 + (c - EMBW)];
  }
  x0[idx] = v;
}

// ---------------- LayerNorm: one wave32 per row -------------------------
__global__ void __launch_bounds__(256)
ln_kernel(const float* __restrict__ x, const float* __restrict__ g,
          const float* __restrict__ b, float* __restrict__ y, int cols) {
  int row  = blockIdx.x * (blockDim.x >> 5) + (threadIdx.x >> 5);
  int lane = threadIdx.x & 31;
  const float* xr = x + (size_t)row * cols;
  float s = 0.f;
  for (int c = lane; c < cols; c += 32) s += xr[c];
  for (int o = 16; o > 0; o >>= 1) s += __shfl_down(s, o);
  float mean = __shfl(s, 0) / (float)cols;
  float v = 0.f;
  for (int c = lane; c < cols; c += 32) { float d = xr[c] - mean; v += d * d; }
  for (int o = 16; o > 0; o >>= 1) v += __shfl_down(v, o);
  float inv = rsqrtf(__shfl(v, 0) / (float)cols + 1e-5f);
  float* yr = y + (size_t)row * cols;
  for (int c = lane; c < cols; c += 32) yr[c] = (xr[c] - mean) * inv * g[c] + b[c];
}

// ---------------- per-head softmax over DH=64 (+scale), wave per chunk ----
__global__ void __launch_bounds__(256)
softmax_head(const float* __restrict__ x, float* __restrict__ y, float scale) {
  int chunk = blockIdx.x * (blockDim.x >> 5) + (threadIdx.x >> 5);
  int lane  = threadIdx.x & 31;
  const float* xr = x + (size_t)chunk * DH;
  float a0 = xr[lane], a1 = xr[lane + 32];
  float mx = fmaxf(a0, a1);
  for (int o = 16; o > 0; o >>= 1) mx = fmaxf(mx, __shfl_down(mx, o));
  mx = __shfl(mx, 0);
  float e0 = expf(a0 - mx), e1 = expf(a1 - mx);
  float s = e0 + e1;
  for (int o = 16; o > 0; o >>= 1) s += __shfl_down(s, o);
  s = __shfl(s, 0);
  float* yr = y + (size_t)chunk * DH;
  yr[lane]      = e0 / s * scale;
  yr[lane + 32] = e1 / s * scale;
}

// ---------------- causal linear attention: block per (b,h), 64 threads ----
// Running state: thread e owns column e of S (64 regs); k,q,Z broadcast in LDS.
__global__ void __launch_bounds__(64)
causal_attn(const float* __restrict__ q, const float* __restrict__ k,
            const float* __restrict__ v, float* __restrict__ o) {
  const int b = blockIdx.x >> 3, h = blockIdx.x & 7;
  const int e = threadIdx.x;
  __shared__ float ksh[DH], qsh[DH], Zsh[DH];
  float Scol[DH];
#pragma unroll
  for (int d = 0; d < DH; ++d) Scol[d] = 0.f;
  Zsh[e] = 0.f;
  const size_t base = ((size_t)b * NSEQ) * DIM + (size_t)h * DH;
  for (int n = 0; n < NSEQ; ++n) {
    const size_t off = base + (size_t)n * DIM;
    __syncthreads();
    ksh[e] = expf(k[off + e]);
    qsh[e] = q[off + e];
    __syncthreads();
    Zsh[e] += ksh[e];
    __syncthreads();
    const float ve = v[off + e];
    float acc = 0.f, D = 0.f;
#pragma unroll 8
    for (int d = 0; d < DH; ++d) {
      Scol[d] += ksh[d] * ve;        // inclusive cumsum of outer(k,v)
      acc     += qsh[d] * Scol[d];
      D       += qsh[d] * Zsh[d];
    }
    o[off + e] = acc / fmaxf(D, 1e-3f);
  }
}

// ---------------- softmax over the sequence axis (cross-attn k) ----------
__global__ void softmax_seq(const float* __restrict__ x, float* __restrict__ y) {
  int idx = blockIdx.x * blockDim.x + threadIdx.x;   // BDIM*DIM
  int b = idx / DIM, c = idx - b * DIM;
  const float* p = x + (size_t)b * NSEQ * DIM + c;
  float mx = -INFINITY;
  for (int n = 0; n < NSEQ; ++n) mx = fmaxf(mx, p[(size_t)n * DIM]);
  float s = 0.f;
  for (int n = 0; n < NSEQ; ++n) s += expf(p[(size_t)n * DIM] - mx);
  float* yp = y + (size_t)b * NSEQ * DIM + c;
  for (int n = 0; n < NSEQ; ++n)
    yp[(size_t)n * DIM] = expf(p[(size_t)n * DIM] - mx) / s;
}

// ---------------- ctx[b,h] = k_s^T @ v  (64x64 per head) -----------------
__global__ void __launch_bounds__(256)
ctx_kernel(const float* __restrict__ ks, const float* __restrict__ v,
           float* __restrict__ ctx) {
  const int bh = blockIdx.x, b = bh >> 3, h = bh & 7;
  const int t = threadIdx.x, d = t >> 2, e0 = (t & 3) << 4;
  float acc[16];
#pragma unroll
  for (int i = 0; i < 16; ++i) acc[i] = 0.f;
  const size_t base = ((size_t)b * NSEQ) * DIM + (size_t)h * DH;
  for (int n = 0; n < NSEQ; ++n) {
    const size_t off = base + (size_t)n * DIM;
    const float kd = ks[off + d];
    const float* vp = v + off + e0;
#pragma unroll
    for (int i = 0; i < 16; ++i) acc[i] += kd * vp[i];
  }
  float* cp = ctx + (size_t)bh * (DH * DH) + (size_t)d * DH + e0;
#pragma unroll
  for (int i = 0; i < 16; ++i) cp[i] = acc[i];
}

// ---------------- out = q_s @ ctx ----------------------------------------
__global__ void apply_ctx(const float* __restrict__ qs,
                          const float* __restrict__ ctx,
                          float* __restrict__ o) {
  int idx = blockIdx.x * blockDim.x + threadIdx.x;   // ROWS*DIM
  int row = idx / DIM, c = idx - row * DIM;
  int h = c >> 6, e = c & 63;
  int b = row / NSEQ;
  const float* cx = ctx + ((size_t)(b * HEADS + h)) * (DH * DH);
  const float* qr = qs + (size_t)row * DIM + h * DH;
  float s = 0.f;
#pragma unroll 8
  for (int d = 0; d < DH; ++d) s += qr[d] * cx[(size_t)d * DH + e];
  o[idx] = s;
}

// ---------------- GEGLU: gelu(u[:, :2048]) * u[:, 2048:] -----------------
__global__ void geglu_kernel(const float* __restrict__ u, float* __restrict__ y) {
  int idx = blockIdx.x * blockDim.x + threadIdx.x;   // ROWS*FFH
  int r = idx / FFH, c = idx - r * FFH;
  float a = u[(size_t)r * FF2 + c];
  float gg = u[(size_t)r * FF2 + FFH + c];
  y[idx] = 0.5f * a * (1.f + erff(a * 0.70710678118654752f)) * gg;
}

__global__ void gelu_kernel(float* __restrict__ x, int n) {
  int i = blockIdx.x * blockDim.x + threadIdx.x;
  if (i < n) { float a = x[i]; x[i] = 0.5f * a * (1.f + erff(a * 0.70710678118654752f)); }
}

__global__ void avg_kernel(const float* __restrict__ a, const float* __restrict__ b,
                           float* __restrict__ y, int n) {
  int i = blockIdx.x * blockDim.x + threadIdx.x;
  if (i < n) y[i] = 0.5f * (a[i] + b[i]);
}

// ---------------- naive GEMM for tiny head outputs (N=4, 8) --------------
__global__ void gemm_naive_act(const float* __restrict__ A, const float* __restrict__ W,
                               const float* __restrict__ bias, float* __restrict__ C,
                               int M, int N, int K, int act) {
  int idx = blockIdx.x * blockDim.x + threadIdx.x;
  if (idx >= M * N) return;
  int m = idx / N, n = idx - m * N;
  float s = bias ? bias[n] : 0.f;
  const float* a = A + (size_t)m * K;
  for (int kk = 0; kk < K; ++kk) s += a[kk] * W[(size_t)kk * N + n];
  if (act) s = 1.f / (1.f + expf(-s));
  C[idx] = s;
}

// ========================================================================
extern "C" void kernel_launch(void* const* d_in, const int* in_sizes, int n_in,
                              void* d_out, int out_size, void* d_ws, size_t ws_size,
                              hipStream_t stream) {
  (void)in_sizes; (void)n_in; (void)out_size; (void)ws_size;
  const float* const* in = (const float* const*)d_in;

  // ---- param indices (setup_inputs dict insertion order) ----
  const float* P_emb   = in[0];
  const float* P_pre_w = in[1];
  const float* P_pre_b = in[2];
  const float* P_pre_g = in[3];
  const float* P_pre_beta = in[4];
  // layers: 6 x 26 leaves at base 5
  const float* P_post_g = in[161];
  const float* P_post_b = in[162];
  const float* const* CLS = &in[163];   // w1,b1,ln_g,ln_b,w2,b2
  const float* const* COL = &in[169];
  const float* const* POS = &in[175];
  const float* saml    = in[181];
  const float* context = in[182];

  // ---- workspace layout (floats) ----
  float* ws = (float*)d_ws;
  const size_t SZ = (size_t)ROWS * DIM;          // 1M floats
  float* x1  = ws;
  float* x2  = ws + SZ;
  float* h   = ws + 2 * SZ;
  float* q   = ws + 3 * SZ;
  float* k   = ws + 4 * SZ;
  float* v   = ws + 5 * SZ;
  float* ao  = ws + 6 * SZ;
  float* u   = ws + 7 * SZ;                      // ROWS*4096
  float* ffh = u + (size_t)ROWS * FF2;           // ROWS*2048
  float* ctx = ffh + (size_t)ROWS * FFH;         // 64*4096
  float* x0  = u;                                // ROWS*128, reused before layers

  auto gemm = [&](const float* A, const float* W, const float* bias,
                  const float* res, float* C, int N, int K) {
    int blocks = (ROWS / 64) * (N / 64);
    gemm_wmma_bf16<<<blocks, 256, 0, stream>>>(A, W, bias, res, C, ROWS, N, K);
  };
  auto ln = [&](const float* x, const float* g, const float* b, float* y, int cols) {
    ln_kernel<<<ROWS / 8, 256, 0, stream>>>(x, g, b, y, cols);
  };

  // ---- embedding + pre projection + pre LN ----
  build_x0<<<(ROWS * X0W) / 256, 256, 0, stream>>>(saml, P_emb, x0);
  gemm(x0, P_pre_w, P_pre_b, nullptr, h, DIM, X0W);
  ln(h, P_pre_g, P_pre_beta, x1, DIM);
  hipMemcpyAsync(x2, x1, SZ * sizeof(float), hipMemcpyDeviceToDevice, stream);

  const float qscale = 0.125f;   // DH^-0.5

  for (int L = 0; L < 6; ++L) {
    const float* const* LP = &in[5 + 26 * L];
    // leaf order within a layer:
    //  0 sa_ln_g  1 sa_ln_b  2 wq 3 wk 4 wv 5 wo 6 bo
    //  7 saff_ln_g 8 saff_ln_b 9 w1 10 b1 11 w2 12 b2
    // 13 ca_ln_g 14 ca_ln_b 15 wq 16 wk 17 wv 18 wo 19 bo
    // 20 caff_ln_g 21 caff_ln_b 22 w1 23 b1 24 w2 25 b2

    // ---- causal self-attention: x1 += attn(LN(x2)) ----
    ln(x2, LP[0], LP[1], h, DIM);
    gemm(h, LP[2], nullptr, nullptr, q, DIM, DIM);
    gemm(h, LP[3], nullptr, nullptr, k, DIM, DIM);
    gemm(h, LP[4], nullptr, nullptr, v, DIM, DIM);
    softmax_head<<<(ROWS * HEADS) / 8, 256, 0, stream>>>(q, q, qscale);
    causal_attn<<<BDIM * HEADS, 64, 0, stream>>>(q, k, v, ao);
    gemm(ao, LP[5], LP[6], x1, x1, DIM, DIM);

    // ---- FF: x2 += GEGLU-FF(LN(x1)) ----
    ln(x1, LP[7], LP[8], h, DIM);
    gemm(h, LP[9], LP[10], nullptr, u, FF2, DIM);
    geglu_kernel<<<(ROWS * FFH) / 256, 256, 0, stream>>>(u, ffh);
    gemm(ffh, LP[11], LP[12], x2, x2, DIM, FFH);

    // ---- cross attention (non-causal): x1 += attn(LN(x2), context) ----
    ln(x2, LP[13], LP[14], h, DIM);
    gemm(h,       LP[15], nullptr, nullptr, q, DIM, DIM);
    gemm(context, LP[16], nullptr, nullptr, k, DIM, DIM);
    gemm(context, LP[17], nullptr, nullptr, v, DIM, DIM);
    softmax_head<<<(ROWS * HEADS) / 8, 256, 0, stream>>>(q, q, qscale);
    softmax_seq<<<(BDIM * DIM) / 256, 256, 0, stream>>>(k, k);
    ctx_kernel<<<BDIM * HEADS, 256, 0, stream>>>(k, v, ctx);
    apply_ctx<<<(ROWS * DIM) / 256, 256, 0, stream>>>(q, ctx, ao);
    gemm(ao, LP[18], LP[19], x1, x1, DIM, DIM);

    // ---- FF: x2 += GEGLU-FF(LN(x1)) ----
    ln(x1, LP[20], LP[21], h, DIM);
    gemm(h, LP[22], LP[23], nullptr, u, FF2, DIM);
    geglu_kernel<<<(ROWS * FFH) / 256, 256, 0, stream>>>(u, ffh);
    gemm(ffh, LP[24], LP[25], x2, x2, DIM, FFH);
  }

  // ---- post: out = LN((x1+x2)/2) -> kept in h ----
  avg_kernel<<<(int)(SZ / 256), 256, 0, stream>>>(x1, x2, q, (int)SZ);
  ln(q, P_post_g, P_post_b, h, DIM);

  float* out = (float*)d_out;
  float* out_emb = out;                               // (ROWS,384)
  float* out_col = out + (size_t)ROWS * NLAY;         // (ROWS,4)
  float* out_pos = out_col + (size_t)ROWS * 4;        // (ROWS,8)

  // ---- cls head: dout=384 via WMMA ----
  gemm(h, CLS[0], CLS[1], nullptr, u, 2 * DIM, DIM);
  ln(u, CLS[2], CLS[3], u, 2 * DIM);
  gelu_kernel<<<(ROWS * 2 * DIM) / 256, 256, 0, stream>>>(u, ROWS * 2 * DIM);
  gemm(u, CLS[4], CLS[5], nullptr, out_emb, NLAY, 2 * DIM);

  // ---- col head: dout=4, sigmoid ----
  gemm(h, COL[0], COL[1], nullptr, u, 2 * DIM, DIM);
  ln(u, COL[2], COL[3], u, 2 * DIM);
  gelu_kernel<<<(ROWS * 2 * DIM) / 256, 256, 0, stream>>>(u, ROWS * 2 * DIM);
  gemm_naive_act<<<(ROWS * 4 + 255) / 256, 256, 0, stream>>>(
      u, COL[4], COL[5], out_col, ROWS, 4, 2 * DIM, 1);

  // ---- pos head: dout=8, sigmoid ----
  gemm(h, POS[0], POS[1], nullptr, u, 2 * DIM, DIM);
  ln(u, POS[2], POS[3], u, 2 * DIM);
  gelu_kernel<<<(ROWS * 2 * DIM) / 256, 256, 0, stream>>>(u, ROWS * 2 * DIM);
  gemm_naive_act<<<(ROWS * 8 + 255) / 256, 256, 0, stream>>>(
      u, POS[4], POS[5], out_pos, ROWS, 8, 2 * DIM, 1);
}
